// ModelClass_90829968376492
// MI455X (gfx1250) — matrix operand order repaired and checked by reading, not compiled, not run on hardware
//
#include <hip/hip_runtime.h>
#include <hip/hip_bf16.h>

typedef __attribute__((ext_vector_type(16))) __bf16 v16bf;
typedef __attribute__((ext_vector_type(8)))  __bf16 v8bf;
typedef __attribute__((ext_vector_type(8)))  float  v8f;
typedef __attribute__((ext_vector_type(4)))  float  v4f;

#define B_ROWS 65536
#define FEAT   512
#define DDIM   128
#define NCLS   1000
#define CTILES 63            // 63*16 = 1008 padded classes
#define EPSN   1e-8f
#define NEGBIG (-3.4e38f)

// workspace layout (bytes)
#define WS_GBF 0              // 1008*128 bf16 = 258048 B
#define WS_WFP (272*1024)     // 512*128  bf16 = 131072 B
#define WS_NLL (416*1024)     // 65536    f32  = 262144 B

// ---------------------------------------------------------------------------
// g_all = normalize(W_g + b_g) -> bf16 [1008][128], rows >= NCLS zeroed.
// One wave per row.
__global__ __launch_bounds__(128) void prep_g(const float* __restrict__ Wg,
                                              const float* __restrict__ bg,
                                              __bf16* __restrict__ gbf) {
    const int wave = threadIdx.x >> 5;
    const int lane = threadIdx.x & 31;
    const int c    = blockIdx.x * 4 + wave;        // 252 blocks * 4 waves = 1008
    float v[4];
    float ss = 0.f;
    #pragma unroll
    for (int j = 0; j < 4; ++j) {
        const int k = lane + 32 * j;
        const float x = (c < NCLS) ? (Wg[(size_t)c * DDIM + k] + bg[k]) : 0.f;
        v[j] = x;
        ss += x * x;
    }
    #pragma unroll
    for (int m = 1; m < 32; m <<= 1) ss += __shfl_xor(ss, m, 32);
    float rn = 1.f / fmaxf(sqrtf(ss), EPSN);
    if (c >= NCLS) rn = 0.f;                       // padded classes -> 0
    #pragma unroll
    for (int j = 0; j < 4; ++j)
        gbf[(size_t)c * DDIM + lane + 32 * j] = (__bf16)(v[j] * rn);
}

// ---------------------------------------------------------------------------
// Pre-pack W_f [512][128] f32 into WMMA B-operand bf16 layout:
//   wfp[(((nt*16 + kc)*32 + lane)*16 + e)] = Wf[k][n]
//   k = kc*32 + (lane/16)*16 + e ,  n = nt*16 + (lane%16)
__global__ __launch_bounds__(256) void prep_wf(const float* __restrict__ Wf,
                                               __bf16* __restrict__ wfp) {
    const int tid  = blockIdx.x * 256 + threadIdx.x;   // 65536 total
    const int e    = tid & 15;
    const int lane = (tid >> 4) & 31;
    const int kc   = (tid >> 9) & 15;
    const int nt   = tid >> 13;
    const int k = kc * 32 + ((lane >> 4) << 4) + e;
    const int n = nt * 16 + (lane & 15);
    wfp[tid] = (__bf16)(Wf[(size_t)k * DDIM + n]);
}

// ---------------------------------------------------------------------------
// Fused: GEMM1 (+bias, row-normalize) -> GEMM2 -> online logsumexp -> NLL/row.
// 8 waves / block; each wave owns a 16-row strip. 512 blocks cover 65536 rows.
__global__ __launch_bounds__(256) void fused(const float* __restrict__ feat,
                                             const int*   __restrict__ tgt,
                                             const float* __restrict__ bfv_g,
                                             const __bf16* __restrict__ wfp,
                                             const __bf16* __restrict__ gbf,
                                             float* __restrict__ nll) {
    __shared__ __bf16 s_f[8][16][DDIM];            // 32 KB: f tile per wave
    const int wave = threadIdx.x >> 5;
    const int lane = threadIdx.x & 31;
    const int h    = lane >> 4;                    // lane half
    const int ln   = lane & 15;
    const int row0 = (blockIdx.x * 8 + wave) * 16;

    // ---------------- GEMM1: [16 x 512] @ [512 x 128] -> 8 N-tiles ----------
    v8f acc[8] = {};
    const float* arow = feat + (size_t)(row0 + ln) * FEAT;
    for (int kc = 0; kc < 16; ++kc) {
        // A operand: k = kc*32 + h*16 + e   (contiguous, matches packed B)
        v16bf a;
        #pragma unroll
        for (int g = 0; g < 2; ++g) {
            const float* p = arow + kc * 32 + h * 16 + g * 8;
            v4f f0 = *(const v4f*)(p);
            v4f f1 = *(const v4f*)(p + 4);
            #pragma unroll
            for (int j = 0; j < 4; ++j) {
                a[g * 8 + j]     = (__bf16)f0[j];
                a[g * 8 + 4 + j] = (__bf16)f1[j];
            }
        }
        #pragma unroll
        for (int nt = 0; nt < 8; ++nt) {
            v16bf b = *(const v16bf*)(wfp + (((nt * 16 + kc) * 32 + lane) << 4));
            acc[nt] = __builtin_amdgcn_wmma_f32_16x16x32_bf16(
                false, a, false, b, (short)0, acc[nt], false, false);
        }
    }

    // ---------------- bias + row L2-normalize -------------------------------
    float ss[8];
    #pragma unroll
    for (int v = 0; v < 8; ++v) ss[v] = 0.f;
    #pragma unroll
    for (int nt = 0; nt < 8; ++nt) {
        const float bb = bfv_g[nt * 16 + ln];      // bias per D-column (N map)
        #pragma unroll
        for (int v = 0; v < 8; ++v) {
            const float x = acc[nt][v] + bb;
            acc[nt][v] = x;
            ss[v] += x * x;
        }
    }
    #pragma unroll
    for (int v = 0; v < 8; ++v) {                  // reduce over 16 lanes/half
        float s = ss[v];
        s += __shfl_xor(s, 1, 32);
        s += __shfl_xor(s, 2, 32);
        s += __shfl_xor(s, 4, 32);
        s += __shfl_xor(s, 8, 32);
        ss[v] = 1.f / fmaxf(sqrtf(s), EPSN);
    }
    // C/D layout: row M = v + 8*h, col = nt*16 + ln  ->  LDS row-major
    #pragma unroll
    for (int nt = 0; nt < 8; ++nt)
        #pragma unroll
        for (int v = 0; v < 8; ++v)
            s_f[wave][v + 8 * h][nt * 16 + ln] = (__bf16)(acc[nt][v] * ss[v]);
    __syncthreads();

    // ---------------- A operand of GEMM2 from LDS (transpose) ---------------
    v16bf a2[4];
    #pragma unroll
    for (int kc = 0; kc < 4; ++kc) {
        #pragma unroll
        for (int g = 0; g < 2; ++g) {
            v8bf t = *(const v8bf*)(&s_f[wave][ln][kc * 32 + h * 16 + g * 8]);
            #pragma unroll
            for (int j = 0; j < 8; ++j) a2[kc][g * 8 + j] = t[j];
        }
    }

    int tv[8];
    #pragma unroll
    for (int v = 0; v < 8; ++v) tv[v] = tgt[row0 + v + 8 * h];

    // ---------------- GEMM2 + streaming logsumexp ---------------------------
    float mr[8], sr[8], val[8];
    #pragma unroll
    for (int v = 0; v < 8; ++v) { mr[v] = NEGBIG; sr[v] = 0.f; val[v] = 0.f; }

    for (int ct = 0; ct < CTILES; ++ct) {
        const int c = ct * 16 + ln;                // class handled by this lane
        const __bf16* grow = gbf + (size_t)c * DDIM;
        v8f acc2 = {};
        #pragma unroll
        for (int kc = 0; kc < 4; ++kc) {
            // B operand: k = kc*32 + h*16 + e  -> 32B contiguous per lane
            v16bf b = *(const v16bf*)(grow + kc * 32 + h * 16);
            acc2 = __builtin_amdgcn_wmma_f32_16x16x32_bf16(
                false, a2[kc], false, b, (short)0, acc2, false, false);
        }
        const bool valid = (c < NCLS);
        #pragma unroll
        for (int v = 0; v < 8; ++v) {
            const float x  = valid ? acc2[v] : NEGBIG;
            const float nm = fmaxf(mr[v], x);
            sr[v] = sr[v] * __expf(mr[v] - nm) + __expf(x - nm);
            mr[v] = nm;
            if (valid && c == tv[v]) val[v] = acc2[v];
        }
    }

    // ---------------- merge across lanes, emit per-row NLL ------------------
    #pragma unroll
    for (int v = 0; v < 8; ++v) {
        float m = mr[v], s = sr[v], vv = val[v];
        #pragma unroll
        for (int msk = 1; msk < 16; msk <<= 1) {
            const float om = __shfl_xor(m, msk, 32);
            const float os = __shfl_xor(s, msk, 32);
            vv += __shfl_xor(vv, msk, 32);
            const float nm = fmaxf(m, om);
            s = s * __expf(m - nm) + os * __expf(om - nm);
            m = nm;
        }
        if (ln == 0) nll[row0 + v + 8 * h] = (m + __logf(s)) - vv;
    }
}

// ---------------------------------------------------------------------------
__global__ __launch_bounds__(256) void reduce_mean(const float* __restrict__ nll,
                                                   float* __restrict__ out) {
    __shared__ float sm[256];
    float s = 0.f;
    for (int i = threadIdx.x; i < B_ROWS; i += 256) s += nll[i];
    sm[threadIdx.x] = s;
    __syncthreads();
    for (int st = 128; st > 0; st >>= 1) {
        if ((int)threadIdx.x < st) sm[threadIdx.x] += sm[threadIdx.x + st];
        __syncthreads();
    }
    if (threadIdx.x == 0) out[0] = sm[0] / (float)B_ROWS;
}

// ---------------------------------------------------------------------------
extern "C" void kernel_launch(void* const* d_in, const int* in_sizes, int n_in,
                              void* d_out, int out_size, void* d_ws, size_t ws_size,
                              hipStream_t stream) {
    (void)in_sizes; (void)n_in; (void)out_size; (void)ws_size;
    const float* feat = (const float*)d_in[0];
    const int*   tgt  = (const int*)  d_in[1];
    const float* Wf   = (const float*)d_in[2];
    const float* bf   = (const float*)d_in[3];
    const float* Wg   = (const float*)d_in[4];
    const float* bg   = (const float*)d_in[5];

    char*   ws  = (char*)d_ws;
    __bf16* gbf = (__bf16*)(ws + WS_GBF);
    __bf16* wfp = (__bf16*)(ws + WS_WFP);
    float*  nll = (float*) (ws + WS_NLL);

    prep_g <<<252, 128, 0, stream>>>(Wg, bg, gbf);
    prep_wf<<<256, 256, 0, stream>>>(Wf, wfp);
    fused  <<<512, 256, 0, stream>>>(feat, tgt, bf, wfp, gbf, nll);
    reduce_mean<<<1, 256, 0, stream>>>(nll, (float*)d_out);
}
